// TernaryMoEBlock_10806137717333
// MI455X (gfx1250) — matrix operand order repaired and checked by reading, not compiled, not run on hardware
//
#include <hip/hip_runtime.h>
#include <hip/hip_bf16.h>
#include <math.h>

#define D_MODEL 1024
#define N_HEADS 16
#define HEAD_DIM 64
#define D_FF 4096
#define N_EXPERTS 8
#define BATCH 4
#define SEQ 1024
#define NTOK (BATCH * SEQ)
#define PER_EXPERT ((size_t)D_FF * D_MODEL)  // 4194304 (same for up and down)

typedef __attribute__((ext_vector_type(16))) __bf16 v16bf;
typedef __attribute__((ext_vector_type(8)))  float  v8f;

static __device__ inline v8f wmma_bf16(v16bf a, v16bf b, v8f c) {
  // D = A(16x32) * B(32x16) + C, bf16 in / f32 out
  return __builtin_amdgcn_wmma_f32_16x16x32_bf16(false, a, false, b, (short)0, c,
                                                 false, false);
}

static __device__ inline v8f zero8() {
  v8f z = {0.f, 0.f, 0.f, 0.f, 0.f, 0.f, 0.f, 0.f};
  return z;
}

// ---- CDNA5 async global->LDS copy (ASYNCcnt path, ISA 15.18.3 op 98) ----
static __device__ inline void async_load_b128(void* lds_ptr, const void* gptr) {
  unsigned lds_off = (unsigned)(uintptr_t)lds_ptr;
  unsigned long long ga = (unsigned long long)(uintptr_t)gptr;
  asm volatile("global_load_async_to_lds_b128 %0, %1, off"
               :: "v"(lds_off), "v"(ga)
               : "memory");
}
static __device__ inline void wait_async0() {
  asm volatile("s_wait_asynccnt 0x0" ::: "memory");
}
// wave issued 6 new async ops for the *next* buffer; ops retire in order, so
// ASYNCcnt <= 6 guarantees the *current* buffer's data has landed in LDS.
static __device__ inline void wait_async6() {
  asm volatile("s_wait_asynccnt 0x6" ::: "memory");
}

// ---------------- block reduction helper (256 threads) ----------------
static __device__ inline float block_sum_256(float v, float* red) {
  int tid = threadIdx.x;
  red[tid] = v;
  __syncthreads();
  for (int off = 128; off > 0; off >>= 1) {
    if (tid < off) red[tid] += red[tid + off];
    __syncthreads();
  }
  float r = red[0];
  __syncthreads();
  return r;
}

// ---------------- fragment loaders (contiguous k order: lane half h holds k 16h..16h+15, wave32) ----------------
static __device__ inline v16bf load_a_frag(const __bf16* src, int ld) {
  int L = threadIdx.x & 31;
  int m = L & 15;
  int kb = (L < 16) ? 0 : 16;
  v16bf a;
#pragma unroll
  for (int j = 0; j < 16; ++j) {
    a[j]     = src[m * ld + kb + j];
  }
  return a;
}

static __device__ inline v16bf load_b_frag(const __bf16* srcNK, int ld) {
  int L = threadIdx.x & 31;
  int n = L & 15;
  int kb = (L < 16) ? 0 : 16;
  v16bf b;
#pragma unroll
  for (int j = 0; j < 16; ++j) b[j] = srcNK[n * ld + kb + j];
  return b;
}

// ---------------- generic bf16 WMMA GEMM: C = A(MxK) * B(NxK)^T ----------------
// 128x64 block tile, 4 waves, double-buffered async LDS pipeline.
enum { EPI_BIAS_BF16 = 0, EPI_BIAS_RES_F32 = 1, EPI_GELU_BF16 = 2, EPI_DOWN_ACC = 3 };

__global__ __launch_bounds__(128) void k_gemm_bf16(
    const __bf16* __restrict__ A, const __bf16* __restrict__ B,
    int M, int N, int K, int epi,
    const float* __restrict__ bias, const float* __restrict__ resid,
    const float* __restrict__ tokw,
    float* __restrict__ Cf, __bf16* __restrict__ Cbf, int ldc) {
  __shared__ __bf16 As[2][128 * 32];
  __shared__ __bf16 Bs[2][64 * 32];

  int n0 = blockIdx.x * 64;
  int m0 = blockIdx.y * 128;
  int wave = threadIdx.x >> 5;
  int L = threadIdx.x & 31;

  // fixed per-thread staging role: rows ar(+32i), cols acol..acol+7
  int ar = threadIdx.x >> 2;         // 0..31
  int acol = (threadIdx.x & 3) * 8;  // 0,8,16,24
  const __bf16* aSrc = A + (size_t)(m0 + ar) * K + acol;
  const __bf16* bSrc = B + (size_t)(n0 + ar) * K + acol;
  const size_t rstep = (size_t)32 * K;  // 32 rows forward
  __bf16* aD[2] = {&As[0][ar * 32 + acol], &As[1][ar * 32 + acol]};
  __bf16* bD[2] = {&Bs[0][ar * 32 + acol], &Bs[1][ar * 32 + acol]};

  auto issue = [&](int buf, int k0) {
    const __bf16* as_ = aSrc + k0;
    const __bf16* bs_ = bSrc + k0;
#pragma unroll
    for (int i = 0; i < 4; ++i)  // A: 128x32 tile
      async_load_b128(aD[buf] + i * (32 * 32), as_ + i * rstep);
#pragma unroll
    for (int i = 0; i < 2; ++i)  // B: 64x32 tile
      async_load_b128(bD[buf] + i * (32 * 32), bs_ + i * rstep);
  };

  v8f acc[2][4];
#pragma unroll
  for (int s = 0; s < 2; ++s)
#pragma unroll
    for (int nt = 0; nt < 4; ++nt) acc[s][nt] = zero8();

  issue(0, 0);
  int buf = 0;
  for (int k0 = 0; k0 < K; k0 += 32, buf ^= 1) {
    if (k0 + 32 < K) {
      issue(buf ^ 1, k0 + 32);  // overlap next tile copy with this tile's WMMAs
      wait_async6();            // current tile landed (only newest 6 may pend)
    } else {
      wait_async0();
    }
    if (k0 + 64 < K) {  // L2 prefetch two K-steps ahead
      __builtin_prefetch(aSrc + k0 + 64, 0, 1);
      __builtin_prefetch(bSrc + k0 + 64, 0, 1);
    }
    __syncthreads();  // publish LDS tile across waves

    v16bf a0 = load_a_frag(&As[buf][(wave * 32) * 32], 32);
    v16bf a1 = load_a_frag(&As[buf][(wave * 32 + 16) * 32], 32);
#pragma unroll
    for (int nt = 0; nt < 4; ++nt) {
      v16bf b = load_b_frag(&Bs[buf][(nt * 16) * 32], 32);
      acc[0][nt] = wmma_bf16(a0, b, acc[0][nt]);
      acc[1][nt] = wmma_bf16(a1, b, acc[1][nt]);
    }
    __syncthreads();  // reads done before peers refill this buffer
  }

  // epilogue: C layout — VGPR r: lanes 0-15 -> row r, lanes 16-31 -> row r+8
  int cn = L & 15;
  int rb = (L < 16) ? 0 : 8;
#pragma unroll
  for (int s = 0; s < 2; ++s) {
#pragma unroll
    for (int nt = 0; nt < 4; ++nt) {
#pragma unroll
      for (int r = 0; r < 8; ++r) {
        int row = m0 + wave * 32 + s * 16 + rb + r;
        int col = n0 + nt * 16 + cn;
        float v = acc[s][nt][r];
        if (epi == EPI_BIAS_BF16) {
          Cbf[(size_t)row * ldc + col] = (__bf16)(v + bias[col]);
        } else if (epi == EPI_BIAS_RES_F32) {
          Cf[(size_t)row * ldc + col] =
              v + bias[col] + resid[(size_t)row * ldc + col];
        } else if (epi == EPI_GELU_BF16) {
          float g = 0.5f * v * (1.0f + erff(v * 0.70710678118654752f));
          Cbf[(size_t)row * ldc + col] = (__bf16)g;
        } else {  // EPI_DOWN_ACC
          size_t idx = (size_t)row * ldc + col;
          Cf[idx] = Cf[idx] + v * tokw[row];
        }
      }
    }
  }
}

// ---------------- WMMA flash attention ----------------
// grid.x = BATCH*N_HEADS (64), grid.y = SEQ/64 (16); 4 waves, 16 q-rows each.
__global__ __launch_bounds__(128) void k_attention(const __bf16* __restrict__ qkv,
                                                   __bf16* __restrict__ o) {
  int bh = blockIdx.x;
  int b = bh >> 4, h = bh & 15;
  int wave = threadIdx.x >> 5;
  int L = threadIdx.x & 31;
  int qbase = blockIdx.y * 64 + wave * 16;

  __shared__ __bf16 Ks[32 * HEAD_DIM];
  __shared__ __bf16 Vs[32 * HEAD_DIM];
  __shared__ __bf16 Ps[4][16 * 32];

  const size_t rs = 3 * D_MODEL;
  const __bf16* qptr = qkv + (size_t)(b * SEQ) * rs + h * HEAD_DIM;
  const __bf16* kptr = qptr + D_MODEL;
  const __bf16* vptr = qptr + 2 * D_MODEL;

  // preload Q 16x64 as two A fragments (K 0..31 / 32..63)
  int m = L & 15;
  int kb8 = (L < 16) ? 0 : 16;
  const __bf16* qrow = qptr + (size_t)(qbase + m) * rs;
  v16bf aq0, aq1;
#pragma unroll
  for (int j = 0; j < 16; ++j) {
    aq0[j]     = qrow[kb8 + j];
    aq1[j]     = qrow[32 + kb8 + j];
  }

  // fixed per-thread staging role for the 32x64 K/V tiles
  int r0 = threadIdx.x >> 3;         // 0..15
  int c0 = (threadIdx.x & 7) * 8;    // 0..56
  const __bf16* kA0 = kptr + (size_t)r0 * rs + c0;
  const __bf16* kA1 = kptr + (size_t)(r0 + 16) * rs + c0;
  const __bf16* vA0 = vptr + (size_t)r0 * rs + c0;
  const __bf16* vA1 = vptr + (size_t)(r0 + 16) * rs + c0;
  __bf16* kD0 = &Ks[r0 * HEAD_DIM + c0];
  __bf16* kD1 = &Ks[(r0 + 16) * HEAD_DIM + c0];
  __bf16* vD0 = &Vs[r0 * HEAD_DIM + c0];
  __bf16* vD1 = &Vs[(r0 + 16) * HEAD_DIM + c0];
  const size_t adv = (size_t)32 * rs;

  v8f oacc[4];
#pragma unroll
  for (int nt = 0; nt < 4; ++nt) oacc[nt] = zero8();
  float mrow[8], lrow[8];
#pragma unroll
  for (int r = 0; r < 8; ++r) { mrow[r] = -1e30f; lrow[r] = 0.f; }

  int rb = (L < 16) ? 0 : 8;
  int kb16 = (L < 16) ? 0 : 16;

  for (int kblk = 0; kblk < SEQ; kblk += 32) {
    __syncthreads();
    async_load_b128(kD0, kA0);
    async_load_b128(kD1, kA1);
    async_load_b128(vD0, vA0);
    async_load_b128(vD1, vA1);
    kA0 += adv; kA1 += adv; vA0 += adv; vA1 += adv;
    wait_async0();
    __syncthreads();

    // scores: S0 = Q @ K[0:16]^T, S1 = Q @ K[16:32]^T (K-dim 64 -> chained)
    v16bf b0a, b0b, b1a, b1b;
    {
      int n = L & 15;
#pragma unroll
      for (int j = 0; j < 16; ++j) {
        b0a[j] = Ks[n * HEAD_DIM + kb16 + j];
        b0b[j] = Ks[n * HEAD_DIM + 32 + kb16 + j];
        b1a[j] = Ks[(16 + n) * HEAD_DIM + kb16 + j];
        b1b[j] = Ks[(16 + n) * HEAD_DIM + 32 + kb16 + j];
      }
    }
    v8f s0 = zero8(), s1 = zero8();
    s0 = wmma_bf16(aq0, b0a, s0);
    s0 = wmma_bf16(aq1, b0b, s0);
    s1 = wmma_bf16(aq0, b1a, s1);
    s1 = wmma_bf16(aq1, b1b, s1);

    // online softmax per row (rows live across 16-lane halves)
#pragma unroll
    for (int r = 0; r < 8; ++r) {
      float v0 = s0[r] * 0.125f;
      float v1 = s1[r] * 0.125f;
      float mx = fmaxf(v0, v1);
      for (int d = 1; d < 16; d <<= 1) mx = fmaxf(mx, __shfl_xor(mx, d, 32));
      float nm = fmaxf(mrow[r], mx);
      float p0 = __expf(v0 - nm);
      float p1 = __expf(v1 - nm);
      float ssum = p0 + p1;
      for (int d = 1; d < 16; d <<= 1) ssum += __shfl_xor(ssum, d, 32);
      float scale = __expf(mrow[r] - nm);
      lrow[r] = lrow[r] * scale + ssum;
      mrow[r] = nm;
#pragma unroll
      for (int nt = 0; nt < 4; ++nt) oacc[nt][r] = oacc[nt][r] * scale;
      int prow = rb + r;
      Ps[wave][prow * 32 + (L & 15)]      = (__bf16)p0;
      Ps[wave][prow * 32 + 16 + (L & 15)] = (__bf16)p1;
    }
    __syncthreads();

    // O += P(16x32) @ V(32x64)
    v16bf ap = load_a_frag(&Ps[wave][0], 32);
#pragma unroll
    for (int nt = 0; nt < 4; ++nt) {
      int n = nt * 16 + (L & 15);
      v16bf bv;
#pragma unroll
      for (int j = 0; j < 16; ++j) bv[j] = Vs[(kb16 + j) * HEAD_DIM + n];
      oacc[nt] = wmma_bf16(ap, bv, oacc[nt]);
    }
  }

  // normalize and store (heads interleaved -> (B*S, D_MODEL))
#pragma unroll
  for (int nt = 0; nt < 4; ++nt) {
#pragma unroll
    for (int r = 0; r < 8; ++r) {
      int row = qbase + rb + r;
      int col = h * HEAD_DIM + nt * 16 + (L & 15);
      float val = oacc[nt][r] / lrow[r];
      o[(size_t)(b * SEQ + row) * D_MODEL + col] = (__bf16)val;
    }
  }
}

// ---------------- LayerNorm 1 -> bf16 ----------------
__global__ __launch_bounds__(256) void k_ln_bf16(const float* __restrict__ x,
                                                 const float* __restrict__ g,
                                                 const float* __restrict__ b,
                                                 __bf16* __restrict__ out) {
  __shared__ float red[256];
  int t = blockIdx.x;
  const float* xr = x + (size_t)t * D_MODEL;
  float s = 0.f, s2 = 0.f;
  for (int i = threadIdx.x; i < D_MODEL; i += 256) {
    float v = xr[i];
    s += v;
    s2 += v * v;
  }
  float mean = block_sum_256(s, red) * (1.0f / D_MODEL);
  float ms = block_sum_256(s2, red) * (1.0f / D_MODEL);
  float rinv = rsqrtf(ms - mean * mean + 1e-5f);
  __bf16* orow = out + (size_t)t * D_MODEL;
  for (int i = threadIdx.x; i < D_MODEL; i += 256)
    orow[i] = (__bf16)((xr[i] - mean) * rinv * g[i] + b[i]);
}

// ---------------- LN2 + geometric router + RMS-norm (fused) ----------------
__global__ __launch_bounds__(256) void k_ln2_router(
    const float* __restrict__ x, const float* __restrict__ g,
    const float* __restrict__ b, const float* __restrict__ router_w,
    __bf16* __restrict__ xn_rms, float* __restrict__ wE) {
  __shared__ float red[256];
  __shared__ float vsh[D_MODEL];
  int t = blockIdx.x;
  const float* xr = x + (size_t)t * D_MODEL;
  float s = 0.f, s2 = 0.f;
  for (int i = threadIdx.x; i < D_MODEL; i += 256) {
    float v = xr[i];
    s += v;
    s2 += v * v;
  }
  float mean = block_sum_256(s, red) * (1.0f / D_MODEL);
  float ms = block_sum_256(s2, red) * (1.0f / D_MODEL);
  float rinv = rsqrtf(ms - mean * mean + 1e-5f);

  float vs2 = 0.f, dp0 = 0.f, dp1 = 0.f, dp2 = 0.f, dp3 = 0.f;
  for (int i = threadIdx.x; i < D_MODEL; i += 256) {
    float v = (xr[i] - mean) * rinv * g[i] + b[i];
    vsh[i] = v;
    vs2 += v * v;
    dp0 += v * router_w[i];
    dp1 += v * router_w[D_MODEL + i];
    dp2 += v * router_w[2 * D_MODEL + i];
    dp3 += v * router_w[3 * D_MODEL + i];
  }
  float sumsq = block_sum_256(vs2, red);
  float h0 = block_sum_256(dp0, red);
  float h1 = block_sum_256(dp1, red);
  float h2 = block_sum_256(dp2, red);
  float h3 = block_sum_256(dp3, red);

  if (threadIdx.x == 0) {
    float n = sqrtf(h0 * h0 + h1 * h1 + h2 * h2 + h3 * h3);
    n = fmaxf(n, 1e-12f);
    h0 /= n; h1 /= n; h2 /= n; h3 /= n;
    const float is2 = 0.70710678118654752f;
    const float phi = 1.61803398874989485f;
    float n3 = rsqrtf(0.75f + 0.25f * phi * phi);
    float d0 = (h0 - h1) * is2;
    float d1 = (h1 - h2) * is2;
    float d2 = h2;
    float d3 = (-0.5f * h0 + 0.5f * h1 + 0.5f * h2 + 0.5f * phi * h3) * n3;
    int ch = (d0 >= 0.f ? 1 : 0) + (d1 >= 0.f ? 2 : 0) + (d2 >= 0.f ? 4 : 0) +
             (d3 >= 0.f ? 8 : 0);
    int e1 = ch & 7;
    int e2 = ((ch >> 1) + 1) & 7;
    if (e1 == e2) e2 = (e2 + 1) & 7;
    float conf = fminf(fminf(fabsf(d0), fabsf(d1)), fminf(fabsf(d2), fabsf(d3)));
    float w1 = 0.5f + 0.3f / (1.f + __expf(-conf));
    float w2 = 1.f - w1;
    for (int e = 0; e < N_EXPERTS; ++e)
      wE[(size_t)e * NTOK + t] = (e == e1) ? w1 : ((e == e2) ? w2 : 0.f);
  }

  float rms = sqrtf(sumsq * (1.0f / D_MODEL));
  float inv = 1.f / (rms + 1e-8f);
  __syncthreads();
  for (int i = threadIdx.x; i < D_MODEL; i += 256)
    xn_rms[(size_t)t * D_MODEL + i] = (__bf16)(vsh[i] * inv);
}

// ---------------- per-row RMS normalization (bf16 -> bf16) ----------------
__global__ __launch_bounds__(256) void k_rms_rows_bf16(const __bf16* __restrict__ in,
                                                       __bf16* __restrict__ out,
                                                       int width) {
  __shared__ float red[256];
  size_t base = (size_t)blockIdx.x * width;
  float s2 = 0.f;
  for (int i = threadIdx.x; i < width; i += 256) {
    float v = (float)in[base + i];
    s2 += v * v;
  }
  float rms = sqrtf(block_sum_256(s2, red) / (float)width);
  float inv = 1.f / (rms + 1e-8f);
  for (int i = threadIdx.x; i < width; i += 256)
    out[base + i] = (__bf16)((float)in[base + i] * inv);
}

// ---------------- alpha = mean|w| : deterministic two stage ----------------
__global__ __launch_bounds__(256) void k_absmean_partial(const float* __restrict__ w,
                                                         float* __restrict__ partial) {
  __shared__ float red[256];
  size_t base = (size_t)blockIdx.x * 4096;
  float s = 0.f;
  for (int i = threadIdx.x; i < 4096; i += 256) s += fabsf(w[base + i]);
  float total = block_sum_256(s, red);
  if (threadIdx.x == 0) partial[blockIdx.x] = total;
}

__global__ __launch_bounds__(256) void k_absmean_final(const float* __restrict__ pu,
                                                       const float* __restrict__ pd,
                                                       float* __restrict__ alpha) {
  __shared__ float red[256];
  const float* p = (blockIdx.x < 8) ? (pu + (size_t)blockIdx.x * 1024)
                                    : (pd + (size_t)(blockIdx.x - 8) * 1024);
  float s = 0.f;
  for (int i = threadIdx.x; i < 1024; i += 256) s += p[i];
  float total = block_sum_256(s, red);
  if (threadIdx.x == 0) alpha[blockIdx.x] = total / (float)PER_EXPERT;
}

// ---------------- ternary quantize fp32 -> bf16 ----------------
__global__ __launch_bounds__(256) void k_ternary_quant(const float* __restrict__ w,
                                                       const float* __restrict__ alpha,
                                                       int alpha_off,
                                                       __bf16* __restrict__ out,
                                                       size_t total) {
  size_t stride = (size_t)gridDim.x * blockDim.x;
  for (size_t idx = (size_t)blockIdx.x * blockDim.x + threadIdx.x; idx < total;
       idx += stride) {
    int e = (int)(idx / PER_EXPERT);
    float a = alpha[alpha_off + e];
    float v = w[idx];
    float t = (fabsf(v) > 0.5f * a) ? (v > 0.f ? 1.f : -1.f) : 0.f;
    out[idx] = (__bf16)t;
  }
}

// ---------------- fp32 -> bf16 convert ----------------
__global__ __launch_bounds__(256) void k_f32_to_bf16(const float* __restrict__ in,
                                                     __bf16* __restrict__ out,
                                                     size_t n) {
  size_t stride = (size_t)gridDim.x * blockDim.x;
  for (size_t i = (size_t)blockIdx.x * blockDim.x + threadIdx.x; i < n; i += stride)
    out[i] = (__bf16)in[i];
}

// ---------------- fp32 copy (init d_out with residual) ----------------
__global__ __launch_bounds__(256) void k_copy_f32(const float* __restrict__ in,
                                                  float* __restrict__ out, size_t n) {
  size_t stride = (size_t)gridDim.x * blockDim.x;
  for (size_t i = (size_t)blockIdx.x * blockDim.x + threadIdx.x; i < n; i += stride)
    out[i] = in[i];
}

// ================================ host ================================
extern "C" void kernel_launch(void* const* d_in, const int* in_sizes, int n_in,
                              void* d_out, int out_size, void* d_ws, size_t ws_size,
                              hipStream_t stream) {
  (void)in_sizes; (void)n_in; (void)out_size; (void)ws_size;
  const float* x        = (const float*)d_in[0];
  const float* g1       = (const float*)d_in[1];
  const float* b1       = (const float*)d_in[2];
  const float* g2       = (const float*)d_in[3];
  const float* b2       = (const float*)d_in[4];
  const float* in_w     = (const float*)d_in[5];
  const float* in_b     = (const float*)d_in[6];
  const float* out_w    = (const float*)d_in[7];
  const float* out_b    = (const float*)d_in[8];
  const float* router_w = (const float*)d_in[9];
  const float* up_w     = (const float*)d_in[10];
  const float* down_w   = (const float*)d_in[11];
  float* outp = (float*)d_out;

  char* cur = (char*)d_ws;
  auto alloc = [&](size_t bytes) -> char* {
    char* r = cur;
    cur += (bytes + 255) & ~(size_t)255;
    return r;
  };
  __bf16* h1       = (__bf16*)alloc((size_t)NTOK * D_MODEL * 2);
  __bf16* in_w_bf  = (__bf16*)alloc((size_t)3 * D_MODEL * D_MODEL * 2);
  __bf16* out_w_bf = (__bf16*)alloc((size_t)D_MODEL * D_MODEL * 2);
  __bf16* qkv      = (__bf16*)alloc((size_t)NTOK * 3 * D_MODEL * 2);
  __bf16* attn_o   = (__bf16*)alloc((size_t)NTOK * D_MODEL * 2);
  float*  x2       = (float*)alloc((size_t)NTOK * D_MODEL * 4);
  __bf16* xn_rms   = (__bf16*)alloc((size_t)NTOK * D_MODEL * 2);
  float*  wE       = (float*)alloc((size_t)N_EXPERTS * NTOK * 4);
  __bf16* upq      = (__bf16*)alloc((size_t)N_EXPERTS * PER_EXPERT * 2);
  __bf16* downq    = (__bf16*)alloc((size_t)N_EXPERTS * PER_EXPERT * 2);
  __bf16* Hbuf     = (__bf16*)alloc((size_t)NTOK * D_FF * 2);
  __bf16* Hrms     = (__bf16*)alloc((size_t)NTOK * D_FF * 2);
  float*  part_up  = (float*)alloc(8192 * 4);
  float*  part_dn  = (float*)alloc(8192 * 4);
  float*  alphas   = (float*)alloc(64);

  // 1) per-expert alpha and ternarized bf16 weights
  k_absmean_partial<<<8192, 256, 0, stream>>>(up_w, part_up);
  k_absmean_partial<<<8192, 256, 0, stream>>>(down_w, part_dn);
  k_absmean_final<<<16, 256, 0, stream>>>(part_up, part_dn, alphas);
  k_ternary_quant<<<4096, 256, 0, stream>>>(up_w, alphas, 0, upq,
                                            (size_t)N_EXPERTS * PER_EXPERT);
  k_ternary_quant<<<4096, 256, 0, stream>>>(down_w, alphas, 8, downq,
                                            (size_t)N_EXPERTS * PER_EXPERT);
  k_f32_to_bf16<<<2048, 256, 0, stream>>>(in_w, in_w_bf, (size_t)3 * D_MODEL * D_MODEL);
  k_f32_to_bf16<<<2048, 256, 0, stream>>>(out_w, out_w_bf, (size_t)D_MODEL * D_MODEL);

  // 2) LN1 + QKV
  k_ln_bf16<<<NTOK, 256, 0, stream>>>(x, g1, b1, h1);
  k_gemm_bf16<<<dim3(3 * D_MODEL / 64, NTOK / 128), 128, 0, stream>>>(
      h1, in_w_bf, NTOK, 3 * D_MODEL, D_MODEL, EPI_BIAS_BF16, in_b, nullptr,
      nullptr, nullptr, qkv, 3 * D_MODEL);

  // 3) flash attention (WMMA)
  k_attention<<<dim3(BATCH * N_HEADS, SEQ / 64), 128, 0, stream>>>(qkv, attn_o);

  // 4) output projection + residual -> x2
  k_gemm_bf16<<<dim3(D_MODEL / 64, NTOK / 128), 128, 0, stream>>>(
      attn_o, out_w_bf, NTOK, D_MODEL, D_MODEL, EPI_BIAS_RES_F32, out_b, x,
      nullptr, x2, nullptr, D_MODEL);

  // 5) LN2 + router + RMS-norm
  k_ln2_router<<<NTOK, 256, 0, stream>>>(x2, g2, b2, router_w, xn_rms, wE);

  // 6) MoE: out = x2 + sum_e wE * down_e(gelu(up_e(xn_rms)))
  k_copy_f32<<<2048, 256, 0, stream>>>(x2, outp, (size_t)NTOK * D_MODEL);
  for (int e = 0; e < N_EXPERTS; ++e) {
    k_gemm_bf16<<<dim3(D_FF / 64, NTOK / 128), 128, 0, stream>>>(
        xn_rms, upq + (size_t)e * PER_EXPERT, NTOK, D_FF, D_MODEL, EPI_GELU_BF16,
        nullptr, nullptr, nullptr, nullptr, Hbuf, D_FF);
    k_rms_rows_bf16<<<NTOK, 256, 0, stream>>>(Hbuf, Hrms, D_FF);
    k_gemm_bf16<<<dim3(D_MODEL / 64, NTOK / 128), 128, 0, stream>>>(
        Hrms, downq + (size_t)e * PER_EXPERT, NTOK, D_MODEL, D_FF, EPI_DOWN_ACC,
        nullptr, nullptr, wE + (size_t)e * NTOK, outp, nullptr, D_MODEL);
  }
}
